// SharedDefromConv_1494648619511
// MI455X (gfx1250) — matrix operand order repaired and checked by reading, not compiled, not run on hardware
//
#include <hip/hip_runtime.h>

// ---------------------------------------------------------------------------
// SharedDeformConv (DCNv2 x3 dilations) for gfx1250 (MI455X).
//
// Main GEMM per branch: M=O=256, N=B*HW=16384, K=2304  (f16 WMMA, f32 acc)
// Offset/mask GEMM:     M=27->32 (padded), same N, K    (f16 WMMA)
//
// K order permuted to K' = k*256 + c: within a 32-wide K chunk the kernel tap
// k = it>>3 is constant -> one bilinear descriptor per thread per chunk.
//
// Distance-2 software pipeline (exploits in-order VMEM retirement):
//   iteration it:
//     1) B frags (LDS) + A frags (global) for chunk it
//     2) WMMA chain on chunk it        (older gathers already retired)
//     3) convert + store gathers of chunk it+1 into the other LDS buffer
//     4) issue gather loads for chunk it+2 (consumed next iteration)
//     5) one barrier
// so gather latency is hidden by: barrier + frag loads + full WMMA chain.
// ---------------------------------------------------------------------------

typedef __attribute__((ext_vector_type(16))) _Float16 v16h;
typedef __attribute__((ext_vector_type(8)))  _Float16 v8h;
typedef __attribute__((ext_vector_type(8)))  float    v8f;

#define ODIM   256
#define HW     4096
#define KDIM   2304      // 9 * 256 (permuted: K' = k*256 + c)
#define KC     32        // K chunk per WMMA
#define KITERS 72        // 2304/32
#define BN     64        // pixels per block (one image row)
#define SB_STRIDE 40     // halves per B-tile row (32 used + pad; 80B, 16B-mult)
#define SB_BUF (BN * SB_STRIDE)

__device__ __forceinline__ v8f wmma16(v16h a, v16h b, v8f c) {
  return __builtin_amdgcn_wmma_f32_16x16x32_f16(
      false, a, false, b, (short)0, c, false, false);
}

// A fragment (16x32, MxK) from global f16 row-major [row][KDIM].
__device__ __forceinline__ v16h load_afrag(const _Float16* __restrict__ W,
                                           int row0, int kbase, int lane) {
  const int r  = row0 + (lane & 15);
  const int kb = kbase + ((lane >> 4) << 3);
  const _Float16* p = W + r * KDIM + kb;
  v8h lo = *(const v8h*)(p);
  v8h hi = *(const v8h*)(p + 16);
  v16h f;
#pragma unroll
  for (int i = 0; i < 8; ++i) { f[i] = lo[i]; f[i + 8] = hi[i]; }
  return f;
}

// B fragment (32x16, KxN) from LDS tile sB[n][SB_STRIDE].
__device__ __forceinline__ v16h load_bfrag(const _Float16* sB, int n0, int lane) {
  const _Float16* p = sB + (n0 + (lane & 15)) * SB_STRIDE + ((lane >> 4) << 3);
  v8h lo = *(const v8h*)(p);
  v8h hi = *(const v8h*)(p + 16);
  v16h f;
#pragma unroll
  for (int i = 0; i < 8; ++i) { f[i] = lo[i]; f[i + 8] = hi[i]; }
  return f;
}

// Issue the 8 im2col loads of phase-1 chunk itn (tap k constant per chunk).
__device__ __forceinline__ void p1_issue(const float* __restrict__ Xb, int py,
                                         int n, int kk0, int itn,
                                         float g[8], bool& inb) {
  const int k  = itn >> 3;
  const int c0 = ((itn << 5) & 255) + kk0;
  const int ky = k / 3, kx = k - ky * 3;
  const int y = py + ky - 1, xx = n + kx - 1;
  inb = ((unsigned)y < 64u) & ((unsigned)xx < 64u);
  const int off = (min(max(y, 0), 63) << 6) + min(max(xx, 0), 63);
#pragma unroll
  for (int e = 0; e < 8; ++e) g[e] = Xb[((c0 + 4 * e) << 12) + off];
}

// Issue the 32 bilinear gathers of phase-3 chunk itn (one descriptor/thread).
__device__ __forceinline__ void p3_issue(const float* __restrict__ Xb,
                                         const int4* sIdx, const float4* sWgt,
                                         int n, int kk0, int itn,
                                         float g[8][4], float4& ww) {
  const int k  = itn >> 3;
  const int c0 = ((itn << 5) & 255) + kk0;
  const int4 ii = sIdx[k * BN + n];
  ww = sWgt[k * BN + n];
#pragma unroll
  for (int e = 0; e < 8; ++e) {
    const float* xp = Xb + ((c0 + 4 * e) << 12);
    g[e][0] = xp[ii.x]; g[e][1] = xp[ii.y];
    g[e][2] = xp[ii.z]; g[e][3] = xp[ii.w];
  }
}

// f32 -> f16 with K permutation K' = k*256 + c  (src K = c*9 + k).
__global__ __launch_bounds__(256) void cvt_kernel(
    const float* __restrict__ weight, const float* __restrict__ w_om,
    _Float16* __restrict__ w16, _Float16* __restrict__ wom16) {
  const int i = blockIdx.x * 256 + threadIdx.x;
  if (i < ODIM * KDIM) {
    const int row = i / KDIM;
    const int kp  = i - row * KDIM;
    const int k = kp >> 8, c = kp & 255;
    w16[i] = (_Float16)weight[row * KDIM + c * 9 + k];
  }
  if (i < 32 * KDIM) {
    const int row = i / KDIM;
    const int kp  = i - row * KDIM;
    const int k = kp >> 8, c = kp & 255;
    wom16[i] = (row < 27) ? (_Float16)w_om[row * KDIM + c * 9 + k]
                          : (_Float16)0.f;
  }
}

__global__ __launch_bounds__(256) void fused_dcn_kernel(
    const float* __restrict__ x1, const float* __restrict__ x2,
    const float* __restrict__ x3, const float* __restrict__ bias,
    const _Float16* __restrict__ w16, const _Float16* __restrict__ wom16,
    float* __restrict__ out) {
  __shared__ _Float16 sB[2 * SB_BUF];       // 10 KB: double-buffered B tile
  __shared__ float    som[32 * BN];         //  8 KB: offset/mask channels
  __shared__ int4     sIdx[9 * BN];         //  9 KB: bilinear corner indices
  __shared__ float4   sWgt[9 * BN];         //  9 KB: corner weights * mask
  __shared__ float    sBias[ODIM];          //  1 KB

  const int tid  = threadIdx.x;
  const int lane = tid & 31;
  const int wv   = tid >> 5;                // 8 waves
  const int br   = blockIdx.y;              // branch 0..2
  const int dil  = br + 1;
  const float* X = (br == 0) ? x1 : ((br == 1) ? x2 : x3);

  const int nG0 = blockIdx.x * BN;          // block = one image row
  const int b   = nG0 >> 12;
  const int py  = (nG0 >> 6) & 63;
  const float* Xb = X + ((size_t)b << 20);  // b * 256 * 4096

  // flat = tid + 256*e  ->  n = tid&63 (fixed), kk = (tid>>6) + 4*e.
  const int n   = tid & 63;
  const int kk0 = tid >> 6;

  sBias[tid] = bias[tid];

  // ================= Phase 1: offset/mask conv (M=32, dil=1, pad=1) ========
  v8f acc1;
#pragma unroll
  for (int i = 0; i < 8; ++i) acc1[i] = 0.f;
  const int mt1 = wv >> 2, nt1 = wv & 3;

  float g1[8];
  bool inb1 = false;
  // Prologue: build chunk 0 directly, then issue gathers for chunk 1.
  {
    float g0[8];
    bool inb0;
    p1_issue(Xb, py, n, kk0, 0, g0, inb0);
#pragma unroll
    for (int e = 0; e < 8; ++e)
      sB[n * SB_STRIDE + kk0 + 4 * e] = (_Float16)(inb0 ? g0[e] : 0.f);
    p1_issue(Xb, py, n, kk0, 1, g1, inb1);
  }
  __syncthreads();

  for (int it = 0; it < KITERS; ++it) {
    const _Float16* bc = sB + (it & 1) * SB_BUF;
    _Float16* bx = sB + ((it + 1) & 1) * SB_BUF;

    v16h bf = load_bfrag(bc, nt1 * 16, lane);
    v16h af = load_afrag(wom16, mt1 * 16, it * KC, lane);
    acc1 = wmma16(af, bf, acc1);

    if (it + 1 < KITERS) {                  // store pre-gathered chunk it+1
#pragma unroll
      for (int e = 0; e < 8; ++e)
        bx[n * SB_STRIDE + kk0 + 4 * e] = (_Float16)(inb1 ? g1[e] : 0.f);
    }
    if (it + 2 < KITERS)                    // issue gathers for chunk it+2
      p1_issue(Xb, py, n, kk0, it + 2, g1, inb1);
    __syncthreads();
  }

  // Dump om accumulators (D layout: VGPR r -> M=r / r+8 by half-wave).
  {
    const int roff = (lane >> 4) << 3;
    const int col  = lane & 15;
#pragma unroll
    for (int r = 0; r < 8; ++r)
      som[(mt1 * 16 + r + roff) * BN + (nt1 * 16 + col)] = acc1[r];
  }
  __syncthreads();

  // ================= Phase 2: bilinear descriptors ==========================
  for (int task = tid; task < 9 * BN; task += 256) {
    const int k  = task >> 6;
    const int nn = task & 63;
    const float dy = som[(2 * k) * BN + nn];
    const float dx = som[(2 * k + 1) * BN + nn];
    const float mz = som[(18 + k) * BN + nn];
    const float m  = 1.f / (1.f + __expf(-mz));
    const int ky = k / 3, kx = k - ky * 3;
    const float sy = (float)(py - dil + ky * dil) + dy;
    const float sx = (float)(nn - dil + kx * dil) + dx;
    const float y0f = floorf(sy), x0f = floorf(sx);
    const float ly = sy - y0f,    lx = sx - x0f;
    const int y0 = (int)y0f, x0 = (int)x0f;
    const int y1 = y0 + 1,   x1i = x0 + 1;
    const float vy0 = (y0 >= 0 && y0 <= 63) ? 1.f : 0.f;
    const float vy1 = (y1 >= 0 && y1 <= 63) ? 1.f : 0.f;
    const float vx0 = (x0 >= 0 && x0 <= 63) ? 1.f : 0.f;
    const float vx1 = (x1i >= 0 && x1i <= 63) ? 1.f : 0.f;
    const int yi0 = min(max(y0, 0), 63), yi1 = min(max(y1, 0), 63);
    const int xi0 = min(max(x0, 0), 63), xi1 = min(max(x1i, 0), 63);
    int4 ii; float4 ww;
    ii.x = yi0 * 64 + xi0;  ww.x = (1.f - ly) * (1.f - lx) * vy0 * vx0 * m;
    ii.y = yi0 * 64 + xi1;  ww.y = (1.f - ly) * lx * vy0 * vx1 * m;
    ii.z = yi1 * 64 + xi0;  ww.z = ly * (1.f - lx) * vy1 * vx0 * m;
    ii.w = yi1 * 64 + xi1;  ww.w = ly * lx * vy1 * vx1 * m;
    sIdx[k * BN + nn] = ii;
    sWgt[k * BN + nn] = ww;
  }
  __syncthreads();

  // ================= Phase 3: main deformable GEMM (M=256) ==================
  v8f acc[4][2];
#pragma unroll
  for (int mt = 0; mt < 4; ++mt)
#pragma unroll
    for (int nt = 0; nt < 2; ++nt)
#pragma unroll
      for (int i = 0; i < 8; ++i) acc[mt][nt][i] = 0.f;

  const int wm = wv >> 1, wn = wv & 1;

  float g[8][4];
  float4 ww;
  // Prologue: build chunk 0 directly, then issue gathers for chunk 1.
  {
    float g0[8][4];
    float4 w0;
    p3_issue(Xb, sIdx, sWgt, n, kk0, 0, g0, w0);
#pragma unroll
    for (int e = 0; e < 8; ++e)
      sB[n * SB_STRIDE + kk0 + 4 * e] =
          (_Float16)(w0.x * g0[e][0] + w0.y * g0[e][1] +
                     w0.z * g0[e][2] + w0.w * g0[e][3]);
    p3_issue(Xb, sIdx, sWgt, n, kk0, 1, g, ww);
  }
  __syncthreads();

  for (int it = 0; it < KITERS; ++it) {
    const _Float16* bc = sB + (it & 1) * SB_BUF;
    _Float16* bx = sB + ((it + 1) & 1) * SB_BUF;

    // Fragment loads + WMMA chain on chunk it (old gathers already retired).
    v16h bf0 = load_bfrag(bc, wn * 32 + 0,  lane);
    v16h bf1 = load_bfrag(bc, wn * 32 + 16, lane);
    v16h af[4];
#pragma unroll
    for (int mt = 0; mt < 4; ++mt)
      af[mt] = load_afrag(w16, wm * 64 + mt * 16, it * KC, lane);
#pragma unroll
    for (int mt = 0; mt < 4; ++mt) {
      acc[mt][0] = wmma16(af[mt], bf0, acc[mt][0]);
      acc[mt][1] = wmma16(af[mt], bf1, acc[mt][1]);
    }

    if (it + 1 < KITERS) {                  // store pre-gathered chunk it+1
#pragma unroll
      for (int e = 0; e < 8; ++e)
        bx[n * SB_STRIDE + kk0 + 4 * e] =
            (_Float16)(ww.x * g[e][0] + ww.y * g[e][1] +
                       ww.z * g[e][2] + ww.w * g[e][3]);
    }
    if (it + 2 < KITERS)                    // issue gathers for chunk it+2
      p3_issue(Xb, sIdx, sWgt, n, kk0, it + 2, g, ww);
    __syncthreads();
  }

  // ================= Epilogue: bias + store =================================
  const size_t outBase =
      (((size_t)br * 4 + (size_t)b) * ODIM) * HW + (size_t)(py << 6);
  const int roff = (lane >> 4) << 3;
  const int col  = lane & 15;
#pragma unroll
  for (int mt = 0; mt < 4; ++mt)
#pragma unroll
    for (int nt = 0; nt < 2; ++nt)
#pragma unroll
      for (int r = 0; r < 8; ++r) {
        const int o  = wm * 64 + mt * 16 + r + roff;
        const int px = wn * 32 + nt * 16 + col;
        out[outBase + (size_t)o * HW + px] = acc[mt][nt][r] + sBias[o];
      }
}

extern "C" void kernel_launch(void* const* d_in, const int* in_sizes, int n_in,
                              void* d_out, int out_size, void* d_ws, size_t ws_size,
                              hipStream_t stream) {
  const float* x1     = (const float*)d_in[0];
  const float* x2     = (const float*)d_in[1];
  const float* x3     = (const float*)d_in[2];
  const float* weight = (const float*)d_in[3];
  const float* bias   = (const float*)d_in[4];
  const float* w_om   = (const float*)d_in[5];
  float* out = (float*)d_out;

  _Float16* w16   = (_Float16*)d_ws;              // 256*2304 f16 (K-permuted)
  _Float16* wom16 = w16 + ODIM * KDIM;            //  32*2304 f16 (padded)

  cvt_kernel<<<(ODIM * KDIM) / 256, 256, 0, stream>>>(weight, w_om, w16, wom16);

  dim3 grid(HW * 4 / BN, 3);                      // 256 row-blocks x 3 branches
  fused_dcn_kernel<<<grid, 256, 0, stream>>>(x1, x2, x3, bias, w16, wom16, out);
}
